// RegionExtractor_64175401336956
// MI455X (gfx1250) — compile-verified, matrix operand order checked by reading, not run on hardware
//
#include <hip/hip_runtime.h>
#include <hip/hip_bf16.h>

// RegionExtractor: per-image (filter small/low-score boxes, sort by score desc,
// greedy NMS @ IoU 0.3, keep first 100) -> regions [B*100,5], mask [B*100].
// One workgroup per image; sort + NMS fully in LDS. WMMA computes the
// pairwise union denominators (area_i + area_j outer-sum) for the top-256
// suppression bitmask tiles.

#define NBOX   2048
#define NT     1024          // 32 waves
#define TK     256           // head size covered by precomputed bitmask
#define MAXD   100
#define IOU_TH 0.3f
#define ASCALE (1.0f/1024.0f)   // area scale so f16 inputs stay < 640 (fp16-safe)

typedef __attribute__((ext_vector_type(16))) _Float16 v16h;
typedef __attribute__((ext_vector_type(8)))  float    v8f;

__global__ __launch_bounds__(NT)
void nms_region_kernel(const float* __restrict__ boxes,
                       const float* __restrict__ scores,
                       float* __restrict__ regions,
                       float* __restrict__ maskout)
{
    __shared__ float skey[NBOX];                       // 8 KB
    __shared__ float sx1[NBOX], sy1[NBOX];             // 16 KB
    __shared__ float sx2[NBOX], sy2[NBOX];             // 16 KB
    __shared__ unsigned short smask16[TK * (TK / 16)]; // 8 KB: row-major, 16 halfwords/row
    __shared__ unsigned short skeptbits[TK / 16];      // kept bitset over head
    __shared__ short skeptIdx[MAXD];
    __shared__ int s_nvalid, s_kept, s_cursor, s_need_tail;
    __shared__ unsigned s_supp;

    const int img = blockIdx.x;
    const int tid = threadIdx.x;
    const float* bp = boxes + (size_t)img * NBOX * 4;
    const float* sp = scores + (size_t)img * NBOX;

    __builtin_prefetch(bp, 0, 1);   // global_prefetch_b8
    __builtin_prefetch(sp, 0, 1);

    // ---------------- Phase A: load, validity filter, sort key ----------------
    for (int i = tid; i < NBOX; i += NT) {
        float x1 = bp[i * 4 + 0], y1 = bp[i * 4 + 1];
        float x2 = bp[i * 4 + 2], y2 = bp[i * 4 + 3];
        float sc = sp[i];
        bool valid = ((x2 - x1) >= 25.0f) & ((y2 - y1) >= 25.0f) & (sc >= 0.001f);
        skey[i] = valid ? sc : -1.0f;     // valid keys >= 0.001 > 0 > -1
        sx1[i] = x1; sy1[i] = y1; sx2[i] = x2; sy2[i] = y2;
    }
    if (tid == 0) s_nvalid = 0;
    __syncthreads();

    // ---------------- Bitonic sort (descending by key, 4-coord payload) -------
    for (int k = 2; k <= NBOX; k <<= 1) {
        for (int j = k >> 1; j > 0; j >>= 1) {
            for (int i = tid; i < NBOX; i += NT) {
                int ixj = i ^ j;
                if (ixj > i) {
                    bool dir = ((i & k) != 0);          // complemented => descending
                    float ki = skey[i], kj = skey[ixj];
                    if ((ki > kj) == dir) {
                        skey[i] = kj; skey[ixj] = ki;
                        float t;
                        t = sx1[i]; sx1[i] = sx1[ixj]; sx1[ixj] = t;
                        t = sy1[i]; sy1[i] = sy1[ixj]; sy1[ixj] = t;
                        t = sx2[i]; sx2[i] = sx2[ixj]; sx2[ixj] = t;
                        t = sy2[i]; sy2[i] = sy2[ixj]; sy2[ixj] = t;
                    }
                }
            }
            __syncthreads();
        }
    }

    // valid count (valid entries are a prefix after descending sort)
    {
        int cnt = 0;
        for (int i = tid; i < NBOX; i += NT) cnt += (skey[i] >= 0.0f) ? 1 : 0;
        if (cnt) atomicAdd(&s_nvalid, cnt);
    }
    __syncthreads();

    // ---------------- Phase B: top-256 suppression bitmask, WMMA denominators -
    // For tile (ti,tj): D[m][n] = (area_i[m] + area_j[n]) * ASCALE via
    // v_wmma_f32_16x16x32_f16 with A = [a_i*s, 1, 0...], B = [1; a_j*s; 0...].
    {
        const int wave = tid >> 5;
        const int lane = tid & 31;
        const int half = lane >> 4;    // 0: lanes 0-15, 1: lanes 16-31
        const int l    = lane & 15;
        const int NTILE = (TK / 16) * (TK / 16);    // 256 tiles, 8 per wave

        for (int t = wave; t < NTILE; t += (NT / 32)) {
            int ti = t >> 4, tj = t & 15;
            v16h a = {};
            v16h b = {};
            if (half == 0) {
                int im = ti * 16 + l;
                float ai = (sx2[im] - sx1[im]) * (sy2[im] - sy1[im]) * ASCALE;
                a[0] = (_Float16)ai;
                a[1] = (_Float16)1.0f;
                int jm = tj * 16 + l;
                float aj = (sx2[jm] - sx1[jm]) * (sy2[jm] - sy1[jm]) * ASCALE;
                b[0] = (_Float16)1.0f;
                b[1] = (_Float16)aj;
            }
            v8f c = {};
            v8f d = __builtin_amdgcn_wmma_f32_16x16x32_f16(
                false, a, false, b, (short)0, c, false, false);

            // this lane owns column N = l of the tile (both halves)
            int jn = tj * 16 + l;
            float jx1 = sx1[jn], jy1 = sy1[jn], jx2 = sx2[jn], jy2 = sy2[jn];
#pragma unroll
            for (int r = 0; r < 8; ++r) {
                int im = ti * 16 + r + half * 8;    // C/D layout: M = r (+8 hi half)
                float iw = fminf(sx2[im], jx2) - fmaxf(sx1[im], jx1);
                float ih = fminf(sy2[im], jy2) - fmaxf(sy1[im], jy1);
                iw = fmaxf(iw, 0.0f);
                ih = fmaxf(ih, 0.0f);
                float inter = iw * ih;
                float uni   = d[r] * (1.0f / ASCALE) - inter; // area_i+area_j-inter
                bool sup = inter > IOU_TH * uni;              // IoU > 0.3
                unsigned bal = __builtin_amdgcn_ballot_w32(sup);
                if (lane == 0) {
                    smask16[(ti * 16 + r) * (TK / 16) + tj]     = (unsigned short)(bal & 0xFFFFu);
                    smask16[(ti * 16 + r + 8) * (TK / 16) + tj] = (unsigned short)(bal >> 16);
                }
            }
        }
    }
    __syncthreads();

    // ---------------- Phase C: greedy scan (serial head via bitmask) ----------
    if (tid == 0) {
        s_kept = 0; s_cursor = 0; s_need_tail = 0; s_supp = 0;
#pragma unroll
        for (int w = 0; w < TK / 16; ++w) skeptbits[w] = 0;
    }
    while (true) {
        if (tid == 0) {
            int i = s_cursor, kept = s_kept;
            int nv = s_nvalid;
            int headEnd = nv < TK ? nv : TK;
            while (i < headEnd && kept < MAXD) {
                const unsigned short* row = &smask16[i * (TK / 16)];
                unsigned acc = 0;
#pragma unroll
                for (int w = 0; w < TK / 16; ++w)
                    acc |= (unsigned)(row[w] & skeptbits[w]);
                if (acc == 0) {
                    skeptIdx[kept++] = (short)i;
                    skeptbits[i >> 4] |= (unsigned short)(1u << (i & 15));
                }
                ++i;
            }
            s_cursor = i;
            s_kept = kept;
            s_need_tail = (kept < MAXD && i < nv) ? 1 : 0;
            s_supp = 0;
        }
        __syncthreads();
        if (!s_need_tail) break;

        // tail candidate (index >= TK): check against kept list in parallel
        {
            int ci = s_cursor;
            int kept = s_kept;
            float cx1 = sx1[ci], cy1 = sy1[ci], cx2 = sx2[ci], cy2 = sy2[ci];
            float ca = (cx2 - cx1) * (cy2 - cy1);
            unsigned mySup = 0;
            for (int k = tid; k < kept; k += NT) {
                int j = skeptIdx[k];
                float iw = fminf(cx2, sx2[j]) - fmaxf(cx1, sx1[j]);
                float ih = fminf(cy2, sy2[j]) - fmaxf(cy1, sy1[j]);
                iw = fmaxf(iw, 0.0f);
                ih = fmaxf(ih, 0.0f);
                float inter = iw * ih;
                float ja = (sx2[j] - sx1[j]) * (sy2[j] - sy1[j]);
                float uni = ca + ja - inter;
                if (inter > IOU_TH * uni) mySup = 1;
            }
            if (mySup) atomicOr(&s_supp, 1u);
        }
        __syncthreads();
        if (tid == 0) {
            if (s_supp == 0) skeptIdx[s_kept++] = (short)s_cursor;
            s_cursor++;
        }
    }
    __syncthreads();

    // ---------------- Phase D: emit regions [img,x1,y1,x2,y2] + mask ----------
    {
        int kept = s_kept;
        float* reg = regions + (size_t)img * MAXD * 5;
        float* mo  = maskout + (size_t)img * MAXD;
        for (int r = tid; r < MAXD; r += NT) {
            float x1 = 0.f, y1 = 0.f, x2 = 0.f, y2 = 0.f, mk = 0.f;
            if (r < kept) {
                int j = skeptIdx[r];
                x1 = sx1[j]; y1 = sy1[j]; x2 = sx2[j]; y2 = sy2[j];
                mk = 1.0f;
            }
            reg[r * 5 + 0] = (float)img;
            reg[r * 5 + 1] = x1;
            reg[r * 5 + 2] = y1;
            reg[r * 5 + 3] = x2;
            reg[r * 5 + 4] = y2;
            mo[r] = mk;
        }
    }
}

extern "C" void kernel_launch(void* const* d_in, const int* in_sizes, int n_in,
                              void* d_out, int out_size, void* d_ws, size_t ws_size,
                              hipStream_t stream) {
    const float* boxes  = (const float*)d_in[0];   // [B, N, 4] f32
    const float* scores = (const float*)d_in[1];   // [B, N]    f32
    int B = in_sizes[1] / NBOX;                    // 8
    float* regions = (float*)d_out;                           // [B*100, 5]
    float* maskout = (float*)d_out + (size_t)B * MAXD * 5;    // [B*100]
    nms_region_kernel<<<dim3(B), dim3(NT), 0, stream>>>(boxes, scores, regions, maskout);
}
